// MultiHeadAttention_87600152969945
// MI455X (gfx1250) — matrix operand order repaired and checked by reading, not compiled, not run on hardware
//
#include <hip/hip_runtime.h>
#include <math.h>

#define D_MODEL 1024
#define NUM_HEADS 16
#define DEPTH 64
#define BATCH 2
#define SEQ 2048
#define ROWS (BATCH * SEQ) /* 4096 */
#define NKT (SEQ / 64)     /* 32 k-tiles in attention */

typedef __bf16 bf16_t;
typedef unsigned int u32;
typedef __attribute__((ext_vector_type(16))) __bf16 v16bf;
typedef __attribute__((ext_vector_type(2))) __bf16 v2bf;
typedef __attribute__((ext_vector_type(2))) float v2f;
typedef __attribute__((ext_vector_type(8))) float v8f;
typedef __attribute__((ext_vector_type(4))) u32 v4u;

static __device__ __forceinline__ v8f wmma_bf16(v16bf a, v16bf b, v8f c) {
  return __builtin_amdgcn_wmma_f32_16x16x32_bf16(false, a, false, b, (short)0, c,
                                                 false, false);
}

static __device__ __forceinline__ bf16_t to_bf16(float x) { return (bf16_t)x; }

// single v_cvt_pk_bf16_f32 with correct pairing via 2-wide vector convert
static __device__ __forceinline__ u32 pack_bf16(float lo, float hi) {
  v2f f;
  f.x = lo;
  f.y = hi;
  union { v2bf v; u32 u; } t;
  t.v = __builtin_convertvector(f, v2bf);
  return t.u;
}

// ---- CDNA5 async global->LDS copy (ASYNCcnt), 16B granules ------------------
static __device__ __forceinline__ u32 lds_off(const void* p) {
  return (u32)(size_t)p;  // low 32 bits of generic LDS address = LDS byte offset
}
static __device__ __forceinline__ void async_copy_b128(u32 ldsByteOff,
                                                       const void* g) {
  asm volatile("global_load_async_to_lds_b128 %0, %1, off"
               :: "v"(ldsByteOff), "v"(g)
               : "memory");
}
static __device__ __forceinline__ void wait_async() {
  asm volatile("s_wait_asynccnt 0x0" ::: "memory");
}

union FragCast { v4u q[2]; v16bf v; };

// A fragment (16x32 bf16), row-major LDS, 2x ds_load_b128 (lda mult of 8 elems)
static __device__ __forceinline__ v16bf load_frag_a(const bf16_t* lds, int lda,
                                                    int row0, int lane) {
  const int r = lane & 15;
  const int khalf = (lane & 16) ? 8 : 0;
  const bf16_t* p = lds + (row0 + r) * lda + khalf;
  FragCast f;
  f.q[0] = *(const v4u*)(p);
  f.q[1] = *(const v4u*)(p + 16);
  return f.v;
}

// B fragment (32x16 bf16) from pair-major LDS Bp[n*ldp + k/2]; 2x ds_load_b128
static __device__ __forceinline__ v16bf load_frag_b(const u32* ldsPairs, int ldp,
                                                    int col0, int lane, int p0) {
  const int n = col0 + (lane & 15);
  const int pbase = p0 + ((lane & 16) ? 4 : 0);
  const u32* p = ldsPairs + n * ldp + pbase;
  FragCast f;
  f.q[0] = *(const v4u*)(p);
  f.q[1] = *(const v4u*)(p + 8);
  return f.v;
}

#define ASTR 40 /* bf16 elems per A row: 80B, 16B aligned */
#define BLDP 20 /* u32 pairs per B column: 16 + 4 pad     */

// -------- Kernel 1: Y = X*W + b (fp32 in), bf16 out head-split ---------------
// storeT==0: dst[((b*H+h)*S + s)*64 + d]; storeT==1: dst[((b*H+h)*64 + d)*S + s]
__global__ __launch_bounds__(256) void proj_kernel(
    const float* __restrict__ X, const float* __restrict__ W,
    const float* __restrict__ bias, unsigned short* __restrict__ dst, float scale,
    int storeT) {
  __shared__ bf16_t As[2][128 * ASTR];
  __shared__ u32 Bp[2][64 * BLDP];
  const int tid = threadIdx.x;
  const int wave = tid >> 5, lane = tid & 31;
  const int m0 = blockIdx.x * 128;
  const int n0 = blockIdx.y * 64;
  const int mb = (wave & 3) * 32;
  const int nb = (wave >> 2) * 32;

  // hoisted staging bases (per-iteration offsets become inst offsets)
  const int ar = tid >> 3, acq = tid & 7;  // A: row, float4-quad
  const float* gA = &X[(size_t)(m0 + ar) * D_MODEL + 4 * acq];
  const int bp = tid >> 4, bj = tid & 15;  // B: k-pair, col-quad
  const float* gB = &W[(size_t)(2 * bp) * D_MODEL + n0 + 4 * bj];

  auto stage = [&](int buf, int k0) {
    bf16_t* AsN = As[buf];
    u32* BpN = Bp[buf];
#pragma unroll
    for (int it = 0; it < 4; ++it) {
      float4 xv = *(const float4*)(gA + k0 + it * (32 * D_MODEL));
      uint2 t; t.x = pack_bf16(xv.x, xv.y); t.y = pack_bf16(xv.z, xv.w);
      *(uint2*)&AsN[(ar + it * 32) * ASTR + 4 * acq] = t;
    }
    const float* wb = gB + (size_t)k0 * D_MODEL;
    float4 a4 = *(const float4*)wb;
    float4 b4 = *(const float4*)(wb + D_MODEL);
    int nbase = 4 * bj;
    BpN[(nbase + 0) * BLDP + bp] = pack_bf16(a4.x, b4.x);
    BpN[(nbase + 1) * BLDP + bp] = pack_bf16(a4.y, b4.y);
    BpN[(nbase + 2) * BLDP + bp] = pack_bf16(a4.z, b4.z);
    BpN[(nbase + 3) * BLDP + bp] = pack_bf16(a4.w, b4.w);
  };

  v8f c00 = {}, c01 = {}, c10 = {}, c11 = {};
  stage(0, 0);
  __syncthreads();
  for (int i = 0; i < 32; ++i) {
    const int cur = i & 1;
    if (i + 1 < 32) stage(cur ^ 1, (i + 1) * 32);  // overlap with WMMA below
    const bf16_t* AsC = As[cur];
    const u32* BpC = Bp[cur];
    v16bf a0 = load_frag_a(AsC, ASTR, mb, lane);
    v16bf a1 = load_frag_a(AsC, ASTR, mb + 16, lane);
    v16bf b0 = load_frag_b(BpC, BLDP, nb, lane, 0);
    v16bf b1 = load_frag_b(BpC, BLDP, nb + 16, lane, 0);
    c00 = wmma_bf16(a0, b0, c00);
    c01 = wmma_bf16(a0, b1, c01);
    c10 = wmma_bf16(a1, b0, c10);
    c11 = wmma_bf16(a1, b1, c11);
    __syncthreads();
  }
  const int rbase = (lane & 16) ? 8 : 0;
  const int colL = lane & 15;
  const int bb = m0 >> 11, sbase = m0 & (SEQ - 1);  // block never crosses batch
#pragma unroll
  for (int i = 0; i < 2; ++i)
#pragma unroll
    for (int j = 0; j < 2; ++j) {
      v8f c = i ? (j ? c11 : c10) : (j ? c01 : c00);
      int gc = n0 + nb + j * 16 + colL;
      int h = gc >> 6, d = gc & 63;
      float bvv = bias[gc];
#pragma unroll
      for (int r = 0; r < 8; ++r) {
        int s = sbase + mb + i * 16 + rbase + r;
        bf16_t hb = to_bf16((c[r] + bvv) * scale);
        size_t idx = storeT
            ? (((size_t)bb * NUM_HEADS + h) * DEPTH + d) * SEQ + s
            : (((size_t)bb * NUM_HEADS + h) * SEQ + s) * DEPTH + d;
        dst[idx] = *(unsigned short*)&hb;
      }
    }
}

#define QSTR 72 /* bf16 elems: 144B rows, 16B aligned */
#define KLDP 36 /* u32 pairs per column: 32 + 4 pad   */
#define SCSTR 65

// -------- Kernel 2: logits = Qh*Kh^T -> d_out attn region + row stats --------
__global__ __launch_bounds__(256) void attn_logits_kernel(
    const unsigned short* __restrict__ qh, const unsigned short* __restrict__ kh,
    float* __restrict__ attn, float* __restrict__ stats) {
  __shared__ bf16_t Qs[128 * QSTR];
  __shared__ u32 Kp[2][64 * KLDP];
  __shared__ float Sc[128 * SCSTR];
  __shared__ float mrow[128], lrow[128];
  const int tid = threadIdx.x;
  const int wave = tid >> 5, lane = tid & 31;
  const int q0 = blockIdx.x * 128;
  const int bh = blockIdx.y;
  const bf16_t* Q = (const bf16_t*)qh + (size_t)bh * SEQ * DEPTH;
  const bf16_t* K = (const bf16_t*)kh + (size_t)bh * SEQ * DEPTH;
  float* attnBH = attn + (size_t)bh * SEQ * SEQ;

  const int cr = tid >> 3, ccq = tid & 7;  // copy helpers (row, b128-quad)
  const bf16_t* gQ = &Q[(size_t)(q0 + cr) * DEPTH + 8 * ccq];
  const bf16_t* gK = &K[(size_t)cr * DEPTH + 8 * ccq];

  auto issueK = [&](int buf, int k0) {
    u32* KpN = Kp[buf];
#pragma unroll
    for (int it = 0; it < 2; ++it)
      async_copy_b128(lds_off(&KpN[(cr + it * 32) * KLDP + 4 * ccq]),
                      gK + ((size_t)k0 + it * 32) * DEPTH);
  };

  // prologue: Q tile (4x b128/thread) + first K tile, all async
#pragma unroll
  for (int it = 0; it < 4; ++it)
    async_copy_b128(lds_off(&Qs[(cr + it * 32) * QSTR + 8 * ccq]),
                    gQ + (size_t)it * 32 * DEPTH);
  issueK(0, 0);
  if (tid < 128) { mrow[tid] = -__builtin_inff(); lrow[tid] = 0.0f; }
  wait_async();
  __syncthreads();

  const int mb = (wave & 3) * 32;
  const int nb = (wave >> 2) * 32;
  const int rbase = (lane & 16) ? 8 : 0;
  const int colL = lane & 15;
  const int sr = tid >> 4, scq = tid & 15;  // Sc stream helpers
  float* gO = &attnBH[(size_t)(q0 + sr) * SEQ + 4 * scq];

  for (int kt = 0; kt < NKT; ++kt) {
    const int cur = kt & 1;
    const int k0 = kt * 64;
    if (kt + 1 < NKT) issueK(cur ^ 1, k0 + 64);  // overlap with WMMA
    const u32* KpC = Kp[cur];
    v8f c00 = {}, c01 = {}, c10 = {}, c11 = {};
#pragma unroll
    for (int kk = 0; kk < DEPTH; kk += 32) {
      v16bf a0 = load_frag_a(Qs + kk, QSTR, mb, lane);
      v16bf a1 = load_frag_a(Qs + kk, QSTR, mb + 16, lane);
      v16bf b0 = load_frag_b(KpC, KLDP, nb, lane, kk >> 1);
      v16bf b1 = load_frag_b(KpC, KLDP, nb + 16, lane, kk >> 1);
      c00 = wmma_bf16(a0, b0, c00);
      c01 = wmma_bf16(a0, b1, c01);
      c10 = wmma_bf16(a1, b0, c10);
      c11 = wmma_bf16(a1, b1, c11);
    }
#pragma unroll
    for (int i = 0; i < 2; ++i)
#pragma unroll
      for (int j = 0; j < 2; ++j) {
        v8f c = i ? (j ? c11 : c10) : (j ? c01 : c00);
        int col = nb + j * 16 + colL;
#pragma unroll
        for (int r = 0; r < 8; ++r)
          Sc[(mb + i * 16 + rbase + r) * SCSTR + col] = c[r];
      }
    __syncthreads();
    // coalesced float4 stream of raw logits to HBM
#pragma unroll
    for (int it = 0; it < 8; ++it) {
      const float* sp = &Sc[(sr + it * 16) * SCSTR + 4 * scq];
      float4 xv; xv.x = sp[0]; xv.y = sp[1]; xv.z = sp[2]; xv.w = sp[3];
      *(float4*)(gO + (size_t)it * 16 * SEQ + k0) = xv;
    }
    if (tid < 128) {  // online softmax stats (stride 65: conflict-free)
      float mOld = mrow[tid];
      float tmax = -__builtin_inff();
#pragma unroll 8
      for (int j = 0; j < 64; ++j) tmax = fmaxf(tmax, Sc[tid * SCSTR + j]);
      float mNew = fmaxf(mOld, tmax);
      float ssum = 0.0f;
#pragma unroll 8
      for (int j = 0; j < 64; ++j) ssum += __expf(Sc[tid * SCSTR + j] - mNew);
      lrow[tid] = lrow[tid] * __expf(mOld - mNew) + ssum;
      mrow[tid] = mNew;
    }
    wait_async();
    __syncthreads();
  }
  if (tid < 128) {
    size_t ridx = (size_t)bh * SEQ + q0 + tid;
    stats[ridx * 2 + 0] = mrow[tid];
    stats[ridx * 2 + 1] = lrow[tid];
  }
}

#define PSTR 72
#define VLDP 36

// -------- Kernel 3: normalize attn in place + O = P*Vh -----------------------
__global__ __launch_bounds__(256) void attn_av_kernel(
    const unsigned short* __restrict__ vt, const float* __restrict__ stats,
    float* __restrict__ attn, unsigned short* __restrict__ oh) {
  __shared__ bf16_t Ps[128 * PSTR];
  __shared__ u32 Vp[2][64 * VLDP];
  __shared__ float mrow[128], linv[128];
  const int tid = threadIdx.x;
  const int wave = tid >> 5, lane = tid & 31;
  const int q0 = blockIdx.x * 128;
  const int bh = blockIdx.y;
  const bf16_t* Vt = (const bf16_t*)vt + (size_t)bh * DEPTH * SEQ;  // [d][s]
  float* attnBH = attn + (size_t)bh * SEQ * SEQ;

  const int cr = tid >> 3, ccq = tid & 7;
  const bf16_t* gV = &Vt[(size_t)cr * SEQ + 8 * ccq];
  auto issueV = [&](int buf, int k0) {
    u32* VpN = Vp[buf];
#pragma unroll
    for (int it = 0; it < 2; ++it)
      async_copy_b128(lds_off(&VpN[(cr + it * 32) * VLDP + 4 * ccq]),
                      gV + (size_t)it * 32 * SEQ + k0);
  };

  issueV(0, 0);
  if (tid < 128) {
    size_t ridx = (size_t)bh * SEQ + q0 + tid;
    mrow[tid] = stats[ridx * 2 + 0];
    linv[tid] = 1.0f / stats[ridx * 2 + 1];
  }
  wait_async();
  __syncthreads();

  const int mb = (wave & 3) * 32;
  const int nb = (wave >> 2) * 32;
  const int rbase = (lane & 16) ? 8 : 0;
  const int colL = lane & 15;
  const int nr = tid >> 4, ncq = tid & 15;  // normalize helpers
  float* gP = &attnBH[(size_t)(q0 + nr) * SEQ + 4 * ncq];
  bf16_t* lP = &Ps[nr * PSTR + 4 * ncq];
  v8f c00 = {}, c01 = {}, c10 = {}, c11 = {};

  for (int kt = 0; kt < NKT; ++kt) {
    const int cur = kt & 1;
    const int k0 = kt * 64;
    if (kt + 1 < NKT) issueV(cur ^ 1, k0 + 64);  // overlap with exp+WMMA
    // normalize 4 cols/thread: float4 in/out + packed bf16 pair stores
#pragma unroll
    for (int it = 0; it < 8; ++it) {
      float* gp = gP + (size_t)it * 16 * SEQ + k0;
      float4 x = *(const float4*)gp;
      float mr = mrow[nr + it * 16], li = linv[nr + it * 16];
      x.x = __expf(x.x - mr) * li;
      x.y = __expf(x.y - mr) * li;
      x.z = __expf(x.z - mr) * li;
      x.w = __expf(x.w - mr) * li;
      *(float4*)gp = x;  // final softmax output
      uint2 t; t.x = pack_bf16(x.x, x.y); t.y = pack_bf16(x.z, x.w);
      *(uint2*)(lP + it * 16 * PSTR) = t;
    }
    __syncthreads();  // Ps visible
    const u32* VpC = Vp[cur];
#pragma unroll
    for (int kk = 0; kk < 64; kk += 32) {
      v16bf a0 = load_frag_a(Ps + kk, PSTR, mb, lane);
      v16bf a1 = load_frag_a(Ps + kk, PSTR, mb + 16, lane);
      v16bf b0 = load_frag_b(VpC, VLDP, nb, lane, kk >> 1);
      v16bf b1 = load_frag_b(VpC, VLDP, nb + 16, lane, kk >> 1);
      c00 = wmma_bf16(a0, b0, c00);
      c01 = wmma_bf16(a0, b1, c01);
      c10 = wmma_bf16(a1, b0, c10);
      c11 = wmma_bf16(a1, b1, c11);
    }
    wait_async();
    __syncthreads();  // Vp[next] ready, Ps free
  }
#pragma unroll
  for (int i = 0; i < 2; ++i)
#pragma unroll
    for (int j = 0; j < 2; ++j) {
      v8f c = i ? (j ? c11 : c10) : (j ? c01 : c00);
      int d = nb + j * 16 + colL;
#pragma unroll
      for (int r = 0; r < 8; ++r) {
        int row = mb + i * 16 + rbase + r;
        bf16_t hb = to_bf16(c[r]);
        oh[((size_t)bh * SEQ + q0 + row) * DEPTH + d] = *(unsigned short*)&hb;
      }
    }
}

// -------- Kernel 4: out = concat(O) * Wo + bo (fp32 to d_out) ----------------
__global__ __launch_bounds__(256) void outproj_kernel(
    const unsigned short* __restrict__ oh, const float* __restrict__ W,
    const float* __restrict__ bias, float* __restrict__ out) {
  __shared__ bf16_t As[2][128 * ASTR];
  __shared__ u32 Bp[2][64 * BLDP];
  const int tid = threadIdx.x;
  const int wave = tid >> 5, lane = tid & 31;
  const int m0 = blockIdx.x * 128;
  const int n0 = blockIdx.y * 64;
  const int mb = (wave & 3) * 32;
  const int nb = (wave >> 2) * 32;
  const bf16_t* O = (const bf16_t*)oh;
  const int bb = m0 >> 11, sbase = m0 & (SEQ - 1);

  const int orr = tid >> 2, ocq = tid & 3;  // A copy: row, b128-quad
  const bf16_t* gOA =
      &O[(((size_t)bb * NUM_HEADS) * SEQ + sbase + orr) * DEPTH + 8 * ocq];
  const int bp = tid >> 4, bj = tid & 15;
  const float* gB = &W[(size_t)(2 * bp) * D_MODEL + n0 + 4 * bj];

  auto issueA = [&](int buf, int k0) {
    const int h = k0 >> 6, dbase = k0 & 63;  // BK=32 never crosses a head
    bf16_t* AsN = As[buf];
#pragma unroll
    for (int it = 0; it < 2; ++it)
      async_copy_b128(
          lds_off(&AsN[(orr + it * 64) * ASTR + 8 * ocq]),
          gOA + ((size_t)h * SEQ + it * 64) * DEPTH + dbase);
  };
  auto stageB = [&](int buf, int k0) {
    u32* BpN = Bp[buf];
    const float* wb = gB + (size_t)k0 * D_MODEL;
    float4 a4 = *(const float4*)wb;
    float4 b4 = *(const float4*)(wb + D_MODEL);
    int nbase = 4 * bj;
    BpN[(nbase + 0) * BLDP + bp] = pack_bf16(a4.x, b4.x);
    BpN[(nbase + 1) * BLDP + bp] = pack_bf16(a4.y, b4.y);
    BpN[(nbase + 2) * BLDP + bp] = pack_bf16(a4.z, b4.z);
    BpN[(nbase + 3) * BLDP + bp] = pack_bf16(a4.w, b4.w);
  };

  v8f c00 = {}, c01 = {}, c10 = {}, c11 = {};
  issueA(0, 0);
  stageB(0, 0);
  wait_async();
  __syncthreads();
  for (int i = 0; i < 32; ++i) {
    const int cur = i & 1;
    if (i + 1 < 32) { issueA(cur ^ 1, (i + 1) * 32); stageB(cur ^ 1, (i + 1) * 32); }
    const bf16_t* AsC = As[cur];
    const u32* BpC = Bp[cur];
    v16bf a0 = load_frag_a(AsC, ASTR, mb, lane);
    v16bf a1 = load_frag_a(AsC, ASTR, mb + 16, lane);
    v16bf b0 = load_frag_b(BpC, BLDP, nb, lane, 0);
    v16bf b1 = load_frag_b(BpC, BLDP, nb + 16, lane, 0);
    c00 = wmma_bf16(a0, b0, c00);
    c01 = wmma_bf16(a0, b1, c01);
    c10 = wmma_bf16(a1, b0, c10);
    c11 = wmma_bf16(a1, b1, c11);
    wait_async();
    __syncthreads();
  }
  const int rbase = (lane & 16) ? 8 : 0;
  const int colL = lane & 15;
#pragma unroll
  for (int i = 0; i < 2; ++i)
#pragma unroll
    for (int j = 0; j < 2; ++j) {
      v8f c = i ? (j ? c11 : c10) : (j ? c01 : c00);
      int gc = n0 + nb + j * 16 + colL;
      float bvv = bias[gc];
#pragma unroll
      for (int r = 0; r < 8; ++r) {
        int gm = m0 + mb + i * 16 + rbase + r;
        out[(size_t)gm * D_MODEL + gc] = c[r] + bvv;
      }
    }
}

extern "C" void kernel_launch(void* const* d_in, const int* in_sizes, int n_in,
                              void* d_out, int out_size, void* d_ws, size_t ws_size,
                              hipStream_t stream) {
  const float* v = (const float*)d_in[0];
  const float* k = (const float*)d_in[1];
  const float* q = (const float*)d_in[2];
  const float* wq = (const float*)d_in[3];
  const float* bq = (const float*)d_in[4];
  const float* wk = (const float*)d_in[5];
  const float* bk = (const float*)d_in[6];
  const float* wv = (const float*)d_in[7];
  const float* bv = (const float*)d_in[8];
  const float* wo = (const float*)d_in[9];
  const float* bo = (const float*)d_in[10];

  float* out = (float*)d_out;
  float* attn = out + (size_t)BATCH * SEQ * D_MODEL;  // attn region of d_out

  unsigned short* qh = (unsigned short*)d_ws;
  unsigned short* kh = qh + (size_t)ROWS * D_MODEL;
  unsigned short* vt = kh + (size_t)ROWS * D_MODEL;  // V transposed [B,H,64,S]
  unsigned short* oh = vt + (size_t)ROWS * D_MODEL;
  float* stats = (float*)(oh + (size_t)ROWS * D_MODEL);

  dim3 blk(256);
  dim3 gProj(ROWS / 128, D_MODEL / 64);
  dim3 gAttn(SEQ / 128, BATCH * NUM_HEADS);

  proj_kernel<<<gProj, blk, 0, stream>>>(q, wq, bq, qh, 0.125f, 0);
  proj_kernel<<<gProj, blk, 0, stream>>>(k, wk, bk, kh, 1.0f, 0);
  proj_kernel<<<gProj, blk, 0, stream>>>(v, wv, bv, vt, 1.0f, 1);
  attn_logits_kernel<<<gAttn, blk, 0, stream>>>(qh, kh, attn, stats);
  attn_av_kernel<<<gAttn, blk, 0, stream>>>(vt, stats, attn, oh);
  outproj_kernel<<<gProj, blk, 0, stream>>>(oh, wo, bo, out);
}
